// FusedMoE_71141838291125
// MI455X (gfx1250) — matrix operand order repaired and checked by reading, not compiled, not run on hardware
//
#include <hip/hip_runtime.h>
#include <hip/hip_bf16.h>
#include <math.h>

// ---------------------------------------------------------------------------
// Fused MoE for gfx1250 (MI455X): bf16 WMMA grouped GEMM, fp32 accumulate,
// TDM (tensor_load_to_lds) for A-tile staging, packed bf16 conversion.
// ---------------------------------------------------------------------------

typedef __bf16 bf16_t;
typedef __attribute__((ext_vector_type(8)))  __bf16 v8bf;
typedef __attribute__((ext_vector_type(16))) __bf16 v16bf;
typedef __attribute__((ext_vector_type(8)))  float  v8f;

#define T_TOK 1024
#define HDIM  2048
#define NEXP  8
#define IDIM  5632
#define RMAX  2560   // 2048 rows + 8 experts * 63 pad, rounded up

// ---- workspace layout (bytes) ----------------------------------------------
static constexpr size_t WS_COUNTS = 0;                                    // int[8]
static constexpr size_t WS_OFFP   = 64;                                   // int[9]
static constexpr size_t WS_TOPKI  = 128;                                  // int[2048]
static constexpr size_t WS_TOPKW  = WS_TOPKI + 2048 * 4;                  // float[2048]
static constexpr size_t WS_TROWS  = WS_TOPKW + 2048 * 4;                  // int[2048]
static constexpr size_t WS_ROWTOK = WS_TROWS + 2048 * 4;                  // int[RMAX]
static constexpr size_t WS_HSB    = ((WS_ROWTOK + RMAX * 4 + 255) / 256) * 256; // bf16[RMAX*HDIM]
static constexpr size_t WS_ACT    = WS_HSB + (size_t)RMAX * HDIM * 2;     // bf16[RMAX*IDIM]
static constexpr size_t WS_OUTR   = WS_ACT + (size_t)RMAX * IDIM * 2;     // f32 [RMAX*HDIM]

// ---- bf16 conversion helpers -----------------------------------------------
__device__ inline unsigned short bfbits(float f) {   // RNE
    unsigned u = __builtin_bit_cast(unsigned, f);
    unsigned r = u + 0x7FFFu + ((u >> 16) & 1u);
    return (unsigned short)(r >> 16);
}
__device__ inline bf16_t f2bf(float f) {
    unsigned short h = bfbits(f);
    return __builtin_bit_cast(bf16_t, h);
}
// packed pair (x -> low 16, y -> high 16) as one 32-bit word
__device__ inline unsigned pack2bf(float x, float y) {
#if __has_builtin(__builtin_amdgcn_cvt_pk_bf16_f32)
    auto p = __builtin_amdgcn_cvt_pk_bf16_f32(x, y);  // v_cvt_pk_bf16_f32
    return __builtin_bit_cast(unsigned, p);
#else
    return (unsigned)bfbits(x) | ((unsigned)bfbits(y) << 16);
#endif
}

__device__ inline v16bf cat16(v8bf lo, v8bf hi) {
    return __builtin_shufflevector(lo, hi,
        0, 1, 2, 3, 4, 5, 6, 7, 8, 9, 10, 11, 12, 13, 14, 15);
}

// ---- Tensor Data Mover: 64-row x 32-elem bf16 tile -> LDS (stride-40 rows) --
#if __has_builtin(__builtin_amdgcn_tensor_load_to_lds) && \
    __has_builtin(__builtin_amdgcn_s_wait_tensorcnt)
#define USE_TDM 1
typedef unsigned int u32x4 __attribute__((ext_vector_type(4)));
typedef int          i32x8 __attribute__((ext_vector_type(8)));
typedef int          i32x4 __attribute__((ext_vector_type(4)));

__device__ inline void tdm_load_tile_a(const bf16_t* gsrc, unsigned lds_off,
                                       unsigned rowLen, unsigned nRows) {
    const unsigned long long ga = (unsigned long long)(size_t)gsrc;
    u32x4 g0;
    g0[0] = 1u;                                        // count=1 (valid D#)
    g0[1] = lds_off;                                   // lds_addr (bytes)
    g0[2] = (unsigned)(ga & 0xFFFFFFFFu);              // global_addr[31:0]
    g0[3] = (unsigned)((ga >> 32) & 0x01FFFFFFu)       // global_addr[56:32]
          | (2u << 30);                                // type = 2 (image)
    i32x8 g1;
    g1[0] = (int)((1u << 16)                           // data_size = 2 bytes
                | (1u << 20)                           // pad_enable
                | (3u << 22)                           // pad_interval: 16 DW (64B)
                | (3u << 25));                         // pad_amount: 4 DW (16B)
    g1[1] = (int)((rowLen & 0xFFFFu) << 16);           // tensor_dim0[15:0]
    g1[2] = (int)(((rowLen >> 16) & 0xFFFFu)           // tensor_dim0[31:16]
                | ((nRows & 0xFFFFu) << 16));          // tensor_dim1[15:0]
    g1[3] = (int)(((nRows >> 16) & 0xFFFFu)            // tensor_dim1[31:16]
                | (32u << 16));                        // tile_dim0 = 32 elems
    g1[4] = 64;                                        // tile_dim1 = 64, tile_dim2 = 0
    g1[5] = (int)rowLen;                               // tensor_dim0_stride[31:0]
    g1[6] = 0;                                         // stride hi / dim1_stride lo
    g1[7] = 0;
    const i32x4 z4 = {0, 0, 0, 0};                     // 2D tensor: groups 2/3 unused
    const i32x8 z8 = {0, 0, 0, 0, 0, 0, 0, 0};
    __builtin_amdgcn_tensor_load_to_lds(g0, g1, z4, z4, z8, 0);
}
#endif

// ---------------------------------------------------------------------------
// 1) Routing: softmax/top-2 per token + padded per-expert offsets
// ---------------------------------------------------------------------------
__global__ void moe_route(const float* __restrict__ logits,
                          int* counts, int* offP,
                          int* __restrict__ topk_i, float* __restrict__ topk_w) {
    __shared__ int scnt[NEXP];
    const int t = threadIdx.x;                       // 1024 threads == T_TOK
    if (t < NEXP) scnt[t] = 0;
    __syncthreads();

    float l[NEXP];
#pragma unroll
    for (int e = 0; e < NEXP; ++e) l[e] = logits[t * NEXP + e];
    int i0 = 0; float m0 = l[0];
#pragma unroll
    for (int e = 1; e < NEXP; ++e) if (l[e] > m0) { m0 = l[e]; i0 = e; }
    int i1 = -1; float m1 = -INFINITY;
#pragma unroll
    for (int e = 0; e < NEXP; ++e) if (e != i0 && l[e] > m1) { m1 = l[e]; i1 = e; }

    const float w0 = 1.0f / (1.0f + __expf(m1 - m0)); // normalized top-2 softmax
    topk_i[t * 2 + 0] = i0;  topk_i[t * 2 + 1] = i1;
    topk_w[t * 2 + 0] = w0;  topk_w[t * 2 + 1] = 1.0f - w0;

    atomicAdd(&scnt[i0], 1);
    atomicAdd(&scnt[i1], 1);
    __syncthreads();

    if (t == 0) {
        int off = 0;
        for (int e = 0; e < NEXP; ++e) {
            counts[e] = scnt[e];
            offP[e]   = off;
            off += ((scnt[e] + 63) >> 6) << 6;       // pad per-expert to 64 rows
        }
        offP[NEXP] = off;
    }
}

// ---------------------------------------------------------------------------
// 2) Deterministic compaction: per-expert token lists via block scan
// ---------------------------------------------------------------------------
__global__ void moe_compact(const int* __restrict__ topk_i,
                            const int* __restrict__ offP,
                            int* __restrict__ token_rows,
                            int* __restrict__ row_token) {
    __shared__ int sc[T_TOK];
    const int e = blockIdx.x;
    const int t = threadIdx.x;

    int k = -1;
    if      (topk_i[t * 2 + 0] == e) k = 0;
    else if (topk_i[t * 2 + 1] == e) k = 1;
    const int flag = (k >= 0) ? 1 : 0;

    sc[t] = flag;
    __syncthreads();
    for (int off = 1; off < T_TOK; off <<= 1) {      // Hillis-Steele inclusive scan
        int v = (t >= off) ? sc[t - off] : 0;
        __syncthreads();
        sc[t] += v;
        __syncthreads();
    }
    const int cnt    = sc[T_TOK - 1];
    const int base   = offP[e];
    const int padded = offP[e + 1] - base;

    if (flag) {
        const int row = base + sc[t] - 1;            // token-order: deterministic
        token_rows[t * 2 + k] = row;
        row_token[row] = t;
    }
    if (t >= cnt && t < padded) row_token[base + t] = -1;   // dummy pad rows
}

// ---------------------------------------------------------------------------
// 3) Gather + convert hidden rows to bf16 (pad rows -> zero), packed stores
// ---------------------------------------------------------------------------
__global__ void moe_stage_hidden(const float* __restrict__ hidden,
                                 const int* __restrict__ row_token,
                                 const int* __restrict__ offP,
                                 bf16_t* __restrict__ hsb) {
    const int r = blockIdx.x;
    if (r >= offP[NEXP]) return;
    const int tok = row_token[r];
    bf16_t* dst = hsb + (size_t)r * HDIM;
    if (tok < 0) {
        for (int i = 0; i < HDIM / 512; ++i)
            *(unsigned*)&dst[(threadIdx.x + i * 256) * 2] = 0u;
    } else {
        const float* src = hidden + (size_t)tok * HDIM;
        for (int i = 0; i < HDIM / 512; ++i) {
            const int h = (threadIdx.x + i * 256) * 2;
            *(unsigned*)&dst[h] = pack2bf(src[h], src[h + 1]);
        }
    }
}

// ---------------------------------------------------------------------------
// 4) Grouped GEMM1: act = silu(hsb @ w13_gate^T) * (hsb @ w13_up^T)
//    block tile M=64 x N=64 gate + paired 64 up cols; 8 waves, 4 WMMA/wave/k
// ---------------------------------------------------------------------------
__global__ __launch_bounds__(256) void moe_gemm1(const bf16_t* __restrict__ hsb,
                                                 const float*  __restrict__ w13,
                                                 const int*    __restrict__ offP,
                                                 bf16_t* __restrict__ act) {
    const int e  = blockIdx.y >> 4;
    const int mb = blockIdx.y & 15;
    const int base   = offP[e];
    const int padded = offP[e + 1] - base;
    const int m0 = mb * 64;
    if (m0 >= padded) return;
    const int rowBase = base + m0;
    const int n0 = blockIdx.x * 64;

    __shared__ __align__(16) bf16_t Al[64][40];      // 64 rows x 32 k (+8 pad)
    __shared__ __align__(16) bf16_t Bg[64][40];      // gate cols x k
    __shared__ __align__(16) bf16_t Bu[64][40];      // up   cols x k

    const int tid  = threadIdx.x;
    const int lane = tid & 31;
    const int wave = tid >> 5;
    const int mt = wave >> 1;                        // wave's 16-row tile (0..3)
    const int nh = wave & 1;                         // wave's 32-col half (0..1)

    const int arow = tid >> 2;                       // staging: 0..63
    const int aseg = tid & 3;                        // staging: 0..3

    const float* gRow = w13 + ((size_t)e * (2 * IDIM) + (n0 + arow)) * HDIM;
    const float* uRow = gRow + (size_t)IDIM * HDIM;
#ifdef USE_TDM
    const unsigned ldsAl = (unsigned)(size_t)&Al[0][0];
#else
    const bf16_t* aRow = hsb + (size_t)(rowBase + arow) * HDIM + aseg * 8;
#endif

    // WMMA fragment coordinates (ISA 7.12.2, wave32, 16-bit A 16x32 / B 32x16)
    const int am = mt * 16 + (lane & 15);
    const int ak = (lane >> 4) * 8;                  // K base 0 or 8 (+16 for hi)
    const int bn = (lane & 15);
    const int bk = (lane >> 4) * 8;                  // K base 0 or 8 (+16 for hi)

    v8f cg0 = {}, cg1 = {}, cu0 = {}, cu1 = {};

    for (int k0 = 0; k0 < HDIM; k0 += 32) {
#ifdef USE_TDM
        if (tid < 32)                                // wave 0 issues one TDM DMA
            tdm_load_tile_a(hsb + (size_t)rowBase * HDIM + k0, ldsAl, HDIM, RMAX);
#else
        *(uint4*)&Al[arow][aseg * 8] = *(const uint4*)(aRow + k0);
#endif
        // stage B gate/up: fp32 load -> packed bf16 convert -> LDS b32 stores
#pragma unroll
        for (int ss = 0; ss < 2; ++ss) {
            const int s = aseg + ss * 4;             // 0..7
            float4 fg = *(const float4*)(gRow + k0 + s * 4);
            float4 fu = *(const float4*)(uRow + k0 + s * 4);
            *(unsigned*)&Bg[arow][s * 4 + 0] = pack2bf(fg.x, fg.y);
            *(unsigned*)&Bg[arow][s * 4 + 2] = pack2bf(fg.z, fg.w);
            *(unsigned*)&Bu[arow][s * 4 + 0] = pack2bf(fu.x, fu.y);
            *(unsigned*)&Bu[arow][s * 4 + 2] = pack2bf(fu.z, fu.w);
        }
        if (k0 + 32 < HDIM) {                        // prefetch next weight slab
            __builtin_prefetch(gRow + k0 + 32 + aseg * 8, 0, 0);
            __builtin_prefetch(uRow + k0 + 32 + aseg * 8, 0, 0);
        }
#ifdef USE_TDM
        __builtin_amdgcn_s_wait_tensorcnt(0);        // TDM tile landed in LDS
#endif
        __syncthreads();

        v16bf a = cat16(*(const v8bf*)&Al[am][ak],
                        *(const v8bf*)&Al[am][ak + 16]);
        v16bf b0g = cat16(*(const v8bf*)&Bg[nh * 32 +  0 + bn][bk],
                          *(const v8bf*)&Bg[nh * 32 +  0 + bn][bk + 16]);
        v16bf b1g = cat16(*(const v8bf*)&Bg[nh * 32 + 16 + bn][bk],
                          *(const v8bf*)&Bg[nh * 32 + 16 + bn][bk + 16]);
        v16bf b0u = cat16(*(const v8bf*)&Bu[nh * 32 +  0 + bn][bk],
                          *(const v8bf*)&Bu[nh * 32 +  0 + bn][bk + 16]);
        v16bf b1u = cat16(*(const v8bf*)&Bu[nh * 32 + 16 + bn][bk],
                          *(const v8bf*)&Bu[nh * 32 + 16 + bn][bk + 16]);

        cg0 = __builtin_amdgcn_wmma_f32_16x16x32_bf16(false, a, false, b0g, (short)0, cg0, false, false);
        cg1 = __builtin_amdgcn_wmma_f32_16x16x32_bf16(false, a, false, b1g, (short)0, cg1, false, false);
        cu0 = __builtin_amdgcn_wmma_f32_16x16x32_bf16(false, a, false, b0u, (short)0, cu0, false, false);
        cu1 = __builtin_amdgcn_wmma_f32_16x16x32_bf16(false, a, false, b1u, (short)0, cu1, false, false);
        __syncthreads();
    }

    // fused SiLU(gate) * up epilogue; C/D layout: VGPR r -> M = r (+8 hi half)
    const int orow = rowBase + mt * 16 + ((lane >> 4) << 3);
    const int ocol = n0 + nh * 32 + (lane & 15);
#pragma unroll
    for (int r = 0; r < 8; ++r) {
        float g0 = cg0[r], g1 = cg1[r];
        float a0 = (g0 / (1.0f + __expf(-g0))) * cu0[r];
        float a1 = (g1 / (1.0f + __expf(-g1))) * cu1[r];
        act[(size_t)(orow + r) * IDIM + ocol]      = f2bf(a0);
        act[(size_t)(orow + r) * IDIM + ocol + 16] = f2bf(a1);
    }
}

// ---------------------------------------------------------------------------
// 5) Grouped GEMM2: out_rows = act @ w2^T   (K = IDIM)
// ---------------------------------------------------------------------------
__global__ __launch_bounds__(256) void moe_gemm2(const bf16_t* __restrict__ act,
                                                 const float*  __restrict__ w2,
                                                 const int*    __restrict__ offP,
                                                 float* __restrict__ outr) {
    const int e  = blockIdx.y >> 4;
    const int mb = blockIdx.y & 15;
    const int base   = offP[e];
    const int padded = offP[e + 1] - base;
    const int m0 = mb * 64;
    if (m0 >= padded) return;
    const int rowBase = base + m0;
    const int n0 = blockIdx.x * 64;                  // over HDIM

    __shared__ __align__(16) bf16_t Al[64][40];
    __shared__ __align__(16) bf16_t Bl[64][40];

    const int tid  = threadIdx.x;
    const int lane = tid & 31;
    const int wave = tid >> 5;
    const int mt = wave >> 1;
    const int nh = wave & 1;

    const int arow = tid >> 2;
    const int aseg = tid & 3;

    const float* bRow = w2 + ((size_t)e * HDIM + (n0 + arow)) * IDIM;
#ifdef USE_TDM
    const unsigned ldsAl = (unsigned)(size_t)&Al[0][0];
#else
    const bf16_t* aRow = act + (size_t)(rowBase + arow) * IDIM + aseg * 8;
#endif

    const int am = mt * 16 + (lane & 15);
    const int ak = (lane >> 4) * 8;
    const int bn = (lane & 15);
    const int bk = (lane >> 4) * 8;

    v8f c0 = {}, c1 = {};

    for (int k0 = 0; k0 < IDIM; k0 += 32) {
#ifdef USE_TDM
        if (tid < 32)
            tdm_load_tile_a(act + (size_t)rowBase * IDIM + k0, ldsAl, IDIM, RMAX);
#else
        *(uint4*)&Al[arow][aseg * 8] = *(const uint4*)(aRow + k0);
#endif
#pragma unroll
        for (int ss = 0; ss < 2; ++ss) {
            const int s = aseg + ss * 4;
            float4 fb = *(const float4*)(bRow + k0 + s * 4);
            *(unsigned*)&Bl[arow][s * 4 + 0] = pack2bf(fb.x, fb.y);
            *(unsigned*)&Bl[arow][s * 4 + 2] = pack2bf(fb.z, fb.w);
        }
        if (k0 + 32 < IDIM)
            __builtin_prefetch(bRow + k0 + 32 + aseg * 8, 0, 0);
#ifdef USE_TDM
        __builtin_amdgcn_s_wait_tensorcnt(0);
#endif
        __syncthreads();

        v16bf a = cat16(*(const v8bf*)&Al[am][ak],
                        *(const v8bf*)&Al[am][ak + 16]);
        v16bf b0 = cat16(*(const v8bf*)&Bl[nh * 32 +  0 + bn][bk],
                         *(const v8bf*)&Bl[nh * 32 +  0 + bn][bk + 16]);
        v16bf b1 = cat16(*(const v8bf*)&Bl[nh * 32 + 16 + bn][bk],
                         *(const v8bf*)&Bl[nh * 32 + 16 + bn][bk + 16]);

        c0 = __builtin_amdgcn_wmma_f32_16x16x32_bf16(false, a, false, b0, (short)0, c0, false, false);
        c1 = __builtin_amdgcn_wmma_f32_16x16x32_bf16(false, a, false, b1, (short)0, c1, false, false);
        __syncthreads();
    }

    const int orow = rowBase + mt * 16 + ((lane >> 4) << 3);
    const int ocol = n0 + nh * 32 + (lane & 15);
#pragma unroll
    for (int r = 0; r < 8; ++r) {
        outr[(size_t)(orow + r) * HDIM + ocol]      = c0[r];
        outr[(size_t)(orow + r) * HDIM + ocol + 16] = c1[r];
    }
}

// ---------------------------------------------------------------------------
// 6) Combine: out[t] = w0 * rows[r0] + w1 * rows[r1]   (deterministic)
// ---------------------------------------------------------------------------
__global__ void moe_combine(const float* __restrict__ out_rows,
                            const int* __restrict__ token_rows,
                            const float* __restrict__ topk_w,
                            float* __restrict__ out) {
    const int t  = blockIdx.x;
    const int r0 = token_rows[t * 2 + 0];
    const int r1 = token_rows[t * 2 + 1];
    const float w0 = topk_w[t * 2 + 0];
    const float w1 = topk_w[t * 2 + 1];
    const float* p0 = out_rows + (size_t)r0 * HDIM;
    const float* p1 = out_rows + (size_t)r1 * HDIM;
    float* po = out + (size_t)t * HDIM;
    for (int i = 0; i < HDIM / 256; ++i) {
        const int h = threadIdx.x + i * 256;
        po[h] = w0 * p0[h] + w1 * p1[h];
    }
}

// ---------------------------------------------------------------------------
extern "C" void kernel_launch(void* const* d_in, const int* in_sizes, int n_in,
                              void* d_out, int out_size, void* d_ws, size_t ws_size,
                              hipStream_t stream) {
    (void)in_sizes; (void)n_in; (void)out_size; (void)ws_size;

    const float* hidden = (const float*)d_in[0];   // (T, H) fp32
    const float* logits = (const float*)d_in[1];   // (T, E) fp32
    const float* w13    = (const float*)d_in[2];   // (E, 2I, H) fp32
    const float* w2     = (const float*)d_in[3];   // (E, H, I) fp32
    // d_in[4] = top_k (== 2, compiled in)

    char* ws = (char*)d_ws;
    int*    counts = (int*)(ws + WS_COUNTS);
    int*    offP   = (int*)(ws + WS_OFFP);
    int*    topki  = (int*)(ws + WS_TOPKI);
    float*  topkw  = (float*)(ws + WS_TOPKW);
    int*    trows  = (int*)(ws + WS_TROWS);
    int*    rowtok = (int*)(ws + WS_ROWTOK);
    bf16_t* hsb    = (bf16_t*)(ws + WS_HSB);
    bf16_t* act    = (bf16_t*)(ws + WS_ACT);
    float*  outr   = (float*)(ws + WS_OUTR);

    moe_route<<<1, T_TOK, 0, stream>>>(logits, counts, offP, topki, topkw);
    moe_compact<<<NEXP, T_TOK, 0, stream>>>(topki, offP, trows, rowtok);
    moe_stage_hidden<<<RMAX, 256, 0, stream>>>(hidden, rowtok, offP, hsb);
    moe_gemm1<<<dim3(IDIM / 64, 16 * NEXP), 256, 0, stream>>>(hsb, w13, offP, act);
    moe_gemm2<<<dim3(HDIM / 64, 16 * NEXP), 256, 0, stream>>>(act, w2, offP, outr);
    moe_combine<<<T_TOK, 256, 0, stream>>>(outr, trows, topkw, (float*)d_out);
}
